// MultiHeadAttention_5901285065103
// MI455X (gfx1250) — matrix operand order repaired and checked by reading, not compiled, not run on hardware
//
#include <hip/hip_runtime.h>
#include <hip/hip_bf16.h>

typedef __attribute__((ext_vector_type(16))) _Float16 v16h;
typedef __attribute__((ext_vector_type(8)))  float    v8f;
typedef __attribute__((ext_vector_type(4)))  int      v4i;

#define DEV __device__ __forceinline__

// ---------------------------------------------------------------------------
// CDNA5 async global->LDS path (guarded; falls back to VGPR staging)
// ---------------------------------------------------------------------------
#if defined(__HIP_DEVICE_COMPILE__) && __has_builtin(__builtin_amdgcn_global_load_async_to_lds_b128)
#define HAVE_ASYNC_LDS 1
#else
#define HAVE_ASYNC_LDS 0
#endif
#if defined(__HIP_DEVICE_COMPILE__) && __has_builtin(__builtin_amdgcn_s_wait_asynccnt)
#define HAVE_ASYNC_WAIT 1
#else
#define HAVE_ASYNC_WAIT 0
#endif

// copy 16 bytes global -> LDS (async DMA on gfx1250: GLOBAL_LOAD_ASYNC_TO_LDS_B128)
DEV void cp16_g2s(_Float16* lds, const _Float16* g) {
#if HAVE_ASYNC_LDS
  __builtin_amdgcn_global_load_async_to_lds_b128(
      (__attribute__((address_space(1))) v4i*)g,
      (__attribute__((address_space(3))) v4i*)lds, 0, 0);
#else
  ((float4*)lds)[0] = ((const float4*)g)[0];
#endif
}

DEV void async_fence() {
#if HAVE_ASYNC_LDS && HAVE_ASYNC_WAIT
  __builtin_amdgcn_s_wait_asynccnt(0);
#endif
}

// ---------------------------------------------------------------------------
// Problem constants
// ---------------------------------------------------------------------------
constexpr int Bb = 8, Ls = 1024, Dm = 1024, Hh = 16, HS = 64, MAXREL = 128;
constexpr int NREL = 2 * MAXREL + 1;              // 257
constexpr size_t ND  = (size_t)Bb * Ls * Dm;      // 8M elements
constexpr size_t NW1 = (size_t)Dm * 3 * Dm;       // 3M
constexpr size_t NW2 = (size_t)Dm * Dm;           // 1M

// workspace offsets (bytes)
constexpr size_t OFF_XH  = 0;
constexpr size_t OFF_WKT = OFF_XH  + ND  * 2;     // W_kqv^T  [3072][1024] f16
constexpr size_t OFF_WOT = OFF_WKT + NW1 * 2;     // W_o^T    [1024][1024] f16
constexpr size_t OFF_QKV = OFF_WOT + NW2 * 2;     // 3 x [B,H,L,HS] f16
constexpr size_t OFF_CTX = OFF_QKV + 3 * ND * 2;  // [B,L,D] f16
constexpr size_t OFF_EK  = OFF_CTX + ND * 2;      // [257][64] f16
constexpr size_t OFF_EV  = OFF_EK  + 33024;
// total ~92.4 MB

// ---------------------------------------------------------------------------
// WMMA helpers (CDNA5 16x16x32 f16 -> f32)
// ---------------------------------------------------------------------------
DEV v8f wmma16(v16h a, v16h b, v8f c) {
  return __builtin_amdgcn_wmma_f32_16x16x32_f16(false, a, false, b,
                                                (short)0, c, false, false);
}

// A fragment: 16x32, M-major tile in LDS (row stride ld halves, ld%8==0).
// lanes 0-15: M=lane, K elems {0..7,16..23}; lanes 16-31: M=lane-16, K {8..15,24..31}
DEV v16h ldsA(const _Float16* s, int row0, int col0, int ld, int lane) {
  const _Float16* p = s + (size_t)(row0 + (lane & 15)) * ld + col0 + ((lane >> 4) << 3);
  v16h a;
  ((float4*)&a)[0] = ((const float4*)p)[0];
  ((float4*)&a)[1] = ((const float4*)(p + 16))[0];
  return a;
}

// B fragment: 32x16, stored N-major in LDS ([N][K], row stride ld halves).
// lanes 0-15: N=lane, K elems {0..7,16..23}; lanes 16-31: N=lane-16, K {8..15,24..31}
DEV v16h ldsB(const _Float16* s, int n0, int col0, int ld, int lane) {
  const _Float16* p = s + (size_t)(n0 + (lane & 15)) * ld + col0 + ((lane >> 4) << 3);
  v16h b;
  ((float4*)&b)[0] = ((const float4*)p)[0];
  ((float4*)&b)[1] = ((const float4*)(p + 16))[0];
  return b;
}

DEV v8f vzero8() { v8f z = {0.f,0.f,0.f,0.f,0.f,0.f,0.f,0.f}; return z; }

// ---------------------------------------------------------------------------
// fp32 -> f16 conversion kernels
// ---------------------------------------------------------------------------
__global__ void k_cvt(const float* __restrict__ src, _Float16* __restrict__ dst, int n) {
  int i = blockIdx.x * 256 + threadIdx.x;
  if (i < n) dst[i] = (_Float16)src[i];
}

// src [rows][cols] fp32 -> dst [cols][rows] f16  (N-major weights)
__global__ void k_cvt_t(const float* __restrict__ src, _Float16* __restrict__ dst,
                        int rows, int cols) {
  int i = blockIdx.x * 256 + threadIdx.x;
  if (i < rows * cols) {
    int r = i / cols, c = i % cols;
    dst[(size_t)c * rows + r] = (_Float16)src[i];
  }
}

// ---------------------------------------------------------------------------
// KQV projection: [8192,1024]f16 @ [1024,3072] (+bias) -> Q/K/V [3][B,H,L,HS]f16
// Tile 128x128x32, 256 threads = 8 waves; wave (wm,wn) owns 32x64.
// ---------------------------------------------------------------------------
__global__ void k_gemm_kqv(const _Float16* __restrict__ Ah,   // [8192][1024]
                           const _Float16* __restrict__ Bt,   // [3072][1024] N-major
                           const float*    __restrict__ bias, // [3072]
                           _Float16*       __restrict__ qkv)  // 3 x [B,H,L,HS]
{
  extern __shared__ char smem[];
  _Float16* As = (_Float16*)smem;        // [128][40]
  _Float16* Bs = As + 128 * 40;          // [128][40]
  const int tid = threadIdx.x, wave = tid >> 5, lane = tid & 31;
  const int n0 = blockIdx.x * 128, m0 = blockIdx.y * 128;
  const int wm = wave >> 1, wn = wave & 1;

  v8f acc[2][4];
  for (int i = 0; i < 2; ++i) for (int j = 0; j < 4; ++j) acc[i][j] = vzero8();

  const int srow = tid >> 1, sseg = (tid & 1) * 16;
  for (int k0 = 0; k0 < 1024; k0 += 32) {
    {
      const _Float16* ga = Ah + (size_t)(m0 + srow) * 1024 + k0 + sseg;
      _Float16* sa = As + srow * 40 + sseg;
      cp16_g2s(sa, ga);
      cp16_g2s(sa + 8, ga + 8);
      const _Float16* gb = Bt + (size_t)(n0 + srow) * 1024 + k0 + sseg;
      _Float16* sb = Bs + srow * 40 + sseg;
      cp16_g2s(sb, gb);
      cp16_g2s(sb + 8, gb + 8);
    }
    async_fence();
    __syncthreads();
    v16h a[2], b[4];
#pragma unroll
    for (int i = 0; i < 2; ++i) a[i] = ldsA(As, wm * 32 + i * 16, 0, 40, lane);
#pragma unroll
    for (int j = 0; j < 4; ++j) b[j] = ldsB(Bs, wn * 64 + j * 16, 0, 40, lane);
#pragma unroll
    for (int i = 0; i < 2; ++i)
#pragma unroll
      for (int j = 0; j < 4; ++j) acc[i][j] = wmma16(a[i], b[j], acc[i][j]);
    __syncthreads();
  }

#pragma unroll
  for (int i = 0; i < 2; ++i)
#pragma unroll
    for (int j = 0; j < 4; ++j)
#pragma unroll
      for (int r = 0; r < 8; ++r) {
        int m = m0 + wm * 32 + i * 16 + r + ((lane >> 4) << 3);
        int n = n0 + wn * 64 + j * 16 + (lane & 15);
        float v = acc[i][j][r] + bias[n];
        int c = n >> 10, d = n & 1023, h = d >> 6, hs = d & 63;
        int bb = m >> 10, l = m & 1023;
        qkv[((((size_t)c * Bb + bb) * Hh + h) * Ls + l) * HS + hs] = (_Float16)v;
      }
}

// ---------------------------------------------------------------------------
// Output projection: ctx[8192,1024]f16 @ W_o (+b_o) -> out fp32
// ---------------------------------------------------------------------------
__global__ void k_gemm_out(const _Float16* __restrict__ Ah,   // [8192][1024]
                           const _Float16* __restrict__ Bt,   // [1024][1024] N-major
                           const float*    __restrict__ bias, // [1024]
                           float*          __restrict__ out)  // [8192][1024] fp32
{
  extern __shared__ char smem[];
  _Float16* As = (_Float16*)smem;
  _Float16* Bs = As + 128 * 40;
  const int tid = threadIdx.x, wave = tid >> 5, lane = tid & 31;
  const int n0 = blockIdx.x * 128, m0 = blockIdx.y * 128;
  const int wm = wave >> 1, wn = wave & 1;

  v8f acc[2][4];
  for (int i = 0; i < 2; ++i) for (int j = 0; j < 4; ++j) acc[i][j] = vzero8();

  const int srow = tid >> 1, sseg = (tid & 1) * 16;
  for (int k0 = 0; k0 < 1024; k0 += 32) {
    {
      const _Float16* ga = Ah + (size_t)(m0 + srow) * 1024 + k0 + sseg;
      _Float16* sa = As + srow * 40 + sseg;
      cp16_g2s(sa, ga);
      cp16_g2s(sa + 8, ga + 8);
      const _Float16* gb = Bt + (size_t)(n0 + srow) * 1024 + k0 + sseg;
      _Float16* sb = Bs + srow * 40 + sseg;
      cp16_g2s(sb, gb);
      cp16_g2s(sb + 8, gb + 8);
    }
    async_fence();
    __syncthreads();
    v16h a[2], b[4];
#pragma unroll
    for (int i = 0; i < 2; ++i) a[i] = ldsA(As, wm * 32 + i * 16, 0, 40, lane);
#pragma unroll
    for (int j = 0; j < 4; ++j) b[j] = ldsB(Bs, wn * 64 + j * 16, 0, 40, lane);
#pragma unroll
    for (int i = 0; i < 2; ++i)
#pragma unroll
      for (int j = 0; j < 4; ++j) acc[i][j] = wmma16(a[i], b[j], acc[i][j]);
    __syncthreads();
  }

#pragma unroll
  for (int i = 0; i < 2; ++i)
#pragma unroll
    for (int j = 0; j < 4; ++j)
#pragma unroll
      for (int r = 0; r < 8; ++r) {
        int m = m0 + wm * 32 + i * 16 + r + ((lane >> 4) << 3);
        int n = n0 + wn * 64 + j * 16 + (lane & 15);
        out[(size_t)m * 1024 + n] = acc[i][j][r] + bias[n];
      }
}

// ---------------------------------------------------------------------------
// Flash attention with Shaw relative positions, all-WMMA.
// Block: (b,h,q-tile of 64). 256 threads / 8 waves. Key tiles of 64.
// ---------------------------------------------------------------------------
constexpr int QLD = 72;    // LDS stride for 64-wide f16 tiles
constexpr int JLD = 136;   // LDS stride for 128-wide f16 tiles
constexpr size_t ATTN_SMEM =
    (size_t)(4608 * 4 /*Qs,Ks,Vst,Ps*/ + 9216 /*EKw*/ + 8704 * 2 /*EVwt,Psh*/) * 2
    + (8192 /*Pw*/ + 4096 /*Ss*/ + 192 /*stats*/ + 512 /*partials*/) * 4;

__global__ void k_attn(const _Float16* __restrict__ Q,
                       const _Float16* __restrict__ K,
                       const _Float16* __restrict__ V,
                       const _Float16* __restrict__ ekh,  // [257][64]
                       const _Float16* __restrict__ evh,  // [257][64]
                       const int*      __restrict__ mask, // [B][L]
                       _Float16*       __restrict__ ctx)  // [B][L][D]
{
  extern __shared__ char smem[];
  _Float16* Qs   = (_Float16*)smem;       // [64][72]  queries
  _Float16* Ks   = Qs   + 64 * QLD;       // [64][72]  keys (N-major for B)
  _Float16* Vst  = Ks   + 64 * QLD;       // [64][72]  V transposed [d][k]
  _Float16* Ps   = Vst  + 64 * QLD;       // [64][72]  softmaxed tile (A for PV)
  _Float16* EKw  = Ps   + 64 * QLD;       // [128][72] rel-k window [j][d]
  _Float16* EVwt = EKw  + 128 * QLD;      // [64][136] rel-v window [d][j]
  _Float16* Psh  = EVwt + 64 * JLD;       // [64][136] diag-scattered P [q][j]
  float* Pw   = (float*)(Psh + 64 * JLD); // [64][128] q . embd_k window
  float* Ss   = Pw + 64 * 128;            // [64][64]  raw scores
  float* m_i  = Ss + 64 * 64;             // [64]
  float* l_i  = m_i + 64;
  float* alph = l_i + 64;
  float* pmax = alph + 64;                // [64][4]
  float* psum = pmax + 256;               // [64][4]

  const int tid = threadIdx.x, wave = tid >> 5, lane = tid & 31;
  const int bh = blockIdx.y;                // b*16 + h
  const int b = bh >> 4, h = bh & 15;
  const int q0 = blockIdx.x * 64;
  const _Float16* Qg = Q + ((size_t)bh * Ls + q0) * HS;
  const _Float16* Kg = K + (size_t)bh * Ls * HS;
  const _Float16* Vg = V + (size_t)bh * Ls * HS;
  const int wm = wave >> 1, wn = wave & 1;  // wave owns M-sub wm, N-subs wn*2+{0,1}

  // stage Q tile once; init stats
  {
    int row = tid >> 2, seg = (tid & 3) * 16;
    const _Float16* g = Qg + row * HS + seg;
    _Float16* s = Qs + row * QLD + seg;
    cp16_g2s(s, g);
    cp16_g2s(s + 8, g + 8);
  }
  if (tid < 64) { m_i[tid] = -3e38f; l_i[tid] = 0.f; }
  v8f O[2]; O[0] = vzero8(); O[1] = vzero8();
  async_fence();
  __syncthreads();

  for (int k0 = 0; k0 < Ls; k0 += 64) {
    const int relbase = k0 - q0 - 63;

    if (k0 + 64 < Ls && tid < 64)
      __builtin_prefetch(Kg + (size_t)(k0 + 64 + tid) * HS, 0, 1);

    // --- stage key tile, V^T, rel windows; zero Psh ---
    {
      int row = tid >> 2, seg = (tid & 3) * 16;
      const _Float16* g = Kg + (size_t)(k0 + row) * HS + seg;
      _Float16* s = Ks + row * QLD + seg;
      cp16_g2s(s, g);
      cp16_g2s(s + 8, g + 8);
    }
    {
      int j = tid >> 1, seg = (tid & 1) * 32;
      int rel = relbase + j;
      rel = rel < -MAXREL ? -MAXREL : (rel > MAXREL ? MAXREL : rel);
      const _Float16* g = ekh + (size_t)(rel + MAXREL) * HS + seg;
      _Float16* s = EKw + j * QLD + seg;
      cp16_g2s(s, g);
      cp16_g2s(s + 8, g + 8);
      cp16_g2s(s + 16, g + 16);
      cp16_g2s(s + 24, g + 24);
    }
    for (int e = tid; e < 64 * 64; e += 256) {
      int d = e >> 6, kk = e & 63;
      Vst[d * QLD + kk] = Vg[(size_t)(k0 + kk) * HS + d];
    }
    for (int e = tid; e < 64 * 128; e += 256) {
      int d = e >> 7, j = e & 127;
      int rel = relbase + j;
      rel = rel < -MAXREL ? -MAXREL : (rel > MAXREL ? MAXREL : rel);
      EVwt[d * JLD + j] = evh[(size_t)(rel + MAXREL) * HS + d];
    }
    for (int e = tid; e < 64 * JLD; e += 256) Psh[e] = (_Float16)0.f;
    async_fence();
    __syncthreads();

    // --- Pw = Q @ EKw^T (64x128, K=64); wave: msub=wm, nsubs=wn*4+{0..3} ---
    {
      v16h a0 = ldsA(Qs, wm * 16, 0, QLD, lane);
      v16h a1 = ldsA(Qs, wm * 16, 32, QLD, lane);
#pragma unroll
      for (int j = 0; j < 4; ++j) {
        int nn = (wn * 4 + j) * 16;
        v16h b0 = ldsB(EKw, nn, 0, QLD, lane);
        v16h b1 = ldsB(EKw, nn, 32, QLD, lane);
        v8f pw = vzero8();
        pw = wmma16(a0, b0, pw);
        pw = wmma16(a1, b1, pw);
#pragma unroll
        for (int r = 0; r < 8; ++r) {
          int m = wm * 16 + r + ((lane >> 4) << 3);
          Pw[m * 128 + nn + (lane & 15)] = pw[r];
        }
      }
      // --- S = Q @ K^T (64x64); wave: msub=wm, nsubs=wn*2+{0,1} ---
#pragma unroll
      for (int j = 0; j < 2; ++j) {
        int nn = (wn * 2 + j) * 16;
        v16h b0 = ldsB(Ks, nn, 0, QLD, lane);
        v16h b1 = ldsB(Ks, nn, 32, QLD, lane);
        v8f s = vzero8();
        s = wmma16(a0, b0, s);
        s = wmma16(a1, b1, s);
#pragma unroll
        for (int r = 0; r < 8; ++r) {
          int m = wm * 16 + r + ((lane >> 4) << 3);
          Ss[m * 64 + nn + (lane & 15)] = s[r];
        }
      }
    }
    __syncthreads();

    // --- online softmax over this tile; 4 threads per query row ---
    const float scale = 0.125f;  // 1/sqrt(64)
    const int row = tid >> 2, c4 = tid & 3;
    float lm = -3e38f;
    for (int t = 0; t < 16; ++t) {
      int n = c4 * 16 + t;
      int j = n - row + 63;                  // rel-window column
      float s = (Ss[row * 64 + n] + Pw[row * 128 + j]) * scale;
      if (mask[b * Ls + k0 + n] == 0) s = -3e38f;
      Ss[row * 64 + n] = s;
      lm = fmaxf(lm, s);
    }
    pmax[row * 4 + c4] = lm;
    __syncthreads();
    if (c4 == 0) {
      float tm = fmaxf(fmaxf(pmax[row * 4], pmax[row * 4 + 1]),
                       fmaxf(pmax[row * 4 + 2], pmax[row * 4 + 3]));
      float mo = m_i[row], mn = fmaxf(mo, tm);
      m_i[row] = mn;
      alph[row] = __expf(mo - mn);
    }
    __syncthreads();
    {
      float mn = m_i[row], ls = 0.f;
      for (int t = 0; t < 16; ++t) {
        int n = c4 * 16 + t;
        float p = __expf(Ss[row * 64 + n] - mn);
        ls += p;
        _Float16 ph = (_Float16)p;
        Ps[row * QLD + n] = ph;
        Psh[row * JLD + (n - row + 63)] = ph;   // diagonal scatter
      }
      psum[row * 4 + c4] = ls;
    }
    __syncthreads();
    if (c4 == 0)
      l_i[row] = l_i[row] * alph[row] + psum[row * 4] + psum[row * 4 + 1] +
                 psum[row * 4 + 2] + psum[row * 4 + 3];
    __syncthreads();

    // --- rescale O, then O += P@V + Psh@EVw^T ---
#pragma unroll
    for (int i = 0; i < 2; ++i)
#pragma unroll
      for (int r = 0; r < 8; ++r)
        O[i][r] *= alph[wm * 16 + r + ((lane >> 4) << 3)];

    {
      v16h a0 = ldsA(Ps, wm * 16, 0, QLD, lane);
      v16h a1 = ldsA(Ps, wm * 16, 32, QLD, lane);
#pragma unroll
      for (int i = 0; i < 2; ++i) {
        int nn = (wn * 2 + i) * 16;
        v16h b0 = ldsB(Vst, nn, 0, QLD, lane);
        v16h b1 = ldsB(Vst, nn, 32, QLD, lane);
        O[i] = wmma16(a0, b0, O[i]);
        O[i] = wmma16(a1, b1, O[i]);
      }
#pragma unroll
      for (int kk = 0; kk < 4; ++kk) {
        v16h a = ldsA(Psh, wm * 16, kk * 32, JLD, lane);
#pragma unroll
        for (int i = 0; i < 2; ++i) {
          int nn = (wn * 2 + i) * 16;
          v16h bfr = ldsB(EVwt, nn, kk * 32, JLD, lane);
          O[i] = wmma16(a, bfr, O[i]);
        }
      }
    }
    __syncthreads();
  }

  // normalize + write ctx f16 [B][L][D]
#pragma unroll
  for (int i = 0; i < 2; ++i)
#pragma unroll
    for (int r = 0; r < 8; ++r) {
      int m = wm * 16 + r + ((lane >> 4) << 3);
      int n = (wn * 2 + i) * 16 + (lane & 15);
      float v = O[i][r] / l_i[m];
      ctx[((size_t)b * Ls + q0 + m) * Dm + h * HS + n] = (_Float16)v;
    }
}

// ---------------------------------------------------------------------------
// Host launcher
// ---------------------------------------------------------------------------
extern "C" void kernel_launch(void* const* d_in, const int* in_sizes, int n_in,
                              void* d_out, int out_size, void* d_ws, size_t ws_size,
                              hipStream_t stream) {
  const float* x    = (const float*)d_in[0];
  const float* Wkqv = (const float*)d_in[1];
  const float* bkqv = (const float*)d_in[2];
  const float* Wo   = (const float*)d_in[3];
  const float* bo   = (const float*)d_in[4];
  const float* ek   = (const float*)d_in[5];
  const float* ev   = (const float*)d_in[6];
  const int*   mask = (const int*)d_in[7];

  char* ws = (char*)d_ws;
  _Float16* xh   = (_Float16*)(ws + OFF_XH);
  _Float16* wkt  = (_Float16*)(ws + OFF_WKT);
  _Float16* wot  = (_Float16*)(ws + OFF_WOT);
  _Float16* qkv  = (_Float16*)(ws + OFF_QKV);
  _Float16* ctxh = (_Float16*)(ws + OFF_CTX);
  _Float16* ekh  = (_Float16*)(ws + OFF_EK);
  _Float16* evh  = (_Float16*)(ws + OFF_EV);

  // precision conversions (+ weight transposes to N-major)
  k_cvt<<<(int)((ND + 255) / 256), 256, 0, stream>>>(x, xh, (int)ND);
  k_cvt_t<<<(int)((NW1 + 255) / 256), 256, 0, stream>>>(Wkqv, wkt, Dm, 3 * Dm);
  k_cvt_t<<<(int)((NW2 + 255) / 256), 256, 0, stream>>>(Wo, wot, Dm, Dm);
  k_cvt<<<(NREL * HS + 255) / 256, 256, 0, stream>>>(ek, ekh, NREL * HS);
  k_cvt<<<(NREL * HS + 255) / 256, 256, 0, stream>>>(ev, evh, NREL * HS);

  // KQV projection
  constexpr size_t GEMM_SMEM = 2 * 128 * 40 * sizeof(_Float16);  // 20480 B
  dim3 g1(3 * Dm / 128, Bb * Ls / 128);  // (24, 64)
  k_gemm_kqv<<<g1, 256, GEMM_SMEM, stream>>>(xh, wkt, bkqv, qkv);

  // attention (Q,K,V chunks are consecutive ND-element slabs of qkv)
  dim3 g2(Ls / 64, Bb * Hh);             // (16, 128)
  k_attn<<<g2, 256, ATTN_SMEM, stream>>>(qkv, qkv + ND, qkv + 2 * ND,
                                         ekh, evh, mask, ctxh);

  // output projection (fp32 out)
  dim3 g3(Dm / 128, Bb * Ls / 128);      // (8, 64)
  k_gemm_out<<<g3, 256, GEMM_SMEM, stream>>>(ctxh, wot, bo, (float*)d_out);
}
